// GraphAttentionLayer_32950989095660
// MI455X (gfx1250) — matrix operand order repaired and checked by reading, not compiled, not run on hardware
//
#include <hip/hip_runtime.h>
#include <math.h>

#define NN   50000
#define EE   800000
#define DIM  256
#define HEADS 4
#define HDIM 64
#define FFD  1024
#define NEG_SLOPE 0.2f
#define LN_EPS 1e-5f

typedef __attribute__((ext_vector_type(16))) _Float16 v16h;
typedef __attribute__((ext_vector_type(8)))  float    v8f;

// ---------------------------------------------------------------------------
// WMMA fragment helpers (V_WMMA_F32_16X16X32_F16, wave32)
// A 16x32 f16: lane L<16 -> row M=L, VGPR r holds K pair at
//   k = (r<4 ? 2r : 2r+8) + 8*(L>=16). Pairs are contiguous -> b128 loads.
// B 32x16 f16 (from transposed weights WT[Nout][K]): lane L -> col N=(L&15),
//   VGPR r holds K pair at k = (r<4 ? 2r : 2r+8) + 8*(L>=16).
// C/D f32: VGPR r -> row = r + 8*(L>=16), col = L&15.
// ---------------------------------------------------------------------------
__device__ inline v16h wmma_load_a(const _Float16* base, int stride, int lane) {
  union { v16h v; unsigned u[8]; } f;
  int m = lane & 15, hi = lane >> 4;
  const _Float16* p = base + (size_t)m * stride + hi * 8;
#pragma unroll
  for (int r = 0; r < 8; ++r) {
    int k = (r < 4) ? (2 * r) : (2 * r + 8);
    f.u[r] = *(const unsigned*)(p + k);
  }
  return f.v;
}

__device__ inline v16h wmma_load_b(const _Float16* wt, int ldk, int n0, int k0, int lane) {
  union { v16h v; unsigned u[8]; } f;
  int j = lane & 15, hi = lane >> 4;
  const _Float16* p = wt + (size_t)(n0 + j) * ldk + k0 + hi * 8;
#pragma unroll
  for (int r = 0; r < 8; ++r)
    f.u[r] = *(const unsigned*)(p + ((r < 4) ? (2 * r) : (2 * r + 8)));
  return f.v;
}

__device__ inline v8f wmma16(v16h a, v16h b, v8f c) {
  return __builtin_amdgcn_wmma_f32_16x16x32_f16(false, a, false, b, (short)0, c, false, false);
}

// ---------------------------------------------------------------------------
// Async global->LDS copy (GLOBAL_LOAD_ASYNC_TO_LDS_B128, tracked by ASYNCcnt)
// VDST = VGPR holding LDS byte address, VADDR = 64-bit global address.
// ---------------------------------------------------------------------------
__device__ inline unsigned lds_addr_of(const void* p) {
  return (unsigned)(unsigned long long)(__attribute__((address_space(3))) const void*)p;
}

__device__ inline void async_copy_b128(void* lds_dst, const void* gsrc) {
  unsigned laddr = lds_addr_of(lds_dst);
  unsigned long long gaddr = (unsigned long long)gsrc;
  asm volatile("global_load_async_to_lds_b128 %0, %1, off"
               :: "v"(laddr), "v"(gaddr) : "memory");
}

__device__ inline void wait_asynccnt0() {
#if __has_builtin(__builtin_amdgcn_s_wait_asynccnt)
  __builtin_amdgcn_s_wait_asynccnt(0);
#else
  asm volatile("s_wait_asynccnt 0" ::: "memory");
#endif
}

__device__ inline void atomicMaxF(float* addr, float v) {
  if (v >= 0.0f) atomicMax((int*)addr, __float_as_int(v));
  else           atomicMin((unsigned int*)addr, __float_as_uint(v));
}

__device__ inline float gelu_exact(float v) {
  return 0.5f * v * (1.0f + erff(v * 0.70710678118654752f));
}

// ---------------------------------------------------------------------------
// K0: convert W[k][n] (fp32, K x Nn) -> WT[n][k] (f16, Nn x K)
// ---------------------------------------------------------------------------
__global__ void k_convert_wT(const float* __restrict__ src, _Float16* __restrict__ dst,
                             int K, int Nn) {
  int gid = blockIdx.x * blockDim.x + threadIdx.x;
  int total = K * Nn;
  if (gid >= total) return;
  int k = gid / Nn, n = gid % Nn;
  dst[(size_t)n * K + k] = (_Float16)src[(size_t)k * Nn + n];
}

// ---------------------------------------------------------------------------
// K_init: zero acc[N*DIM], denom[N*H]; m[N*H] = -inf
// ---------------------------------------------------------------------------
__global__ void k_init(float* __restrict__ acc, float* __restrict__ denom,
                       float* __restrict__ mbuf) {
  int stride = gridDim.x * blockDim.x;
  for (int i = blockIdx.x * blockDim.x + threadIdx.x; i < NN * DIM; i += stride)
    acc[i] = 0.0f;
  for (int i = blockIdx.x * blockDim.x + threadIdx.x; i < NN * HEADS; i += stride) {
    denom[i] = 0.0f;
    mbuf[i] = -__builtin_inff();
  }
}

// ---------------------------------------------------------------------------
// K1: h = x @ W  (WMMA). 16 rows/block, 8 waves x 2 n-tiles.
// B K-slices (256 cols x 32 k = 16 KB f16) are double-buffered in LDS via
// async global->LDS loads (ASYNCcnt pipeline): prefetch slice kt+1 while
// WMMAs consume slice kt.
// ---------------------------------------------------------------------------
__global__ void __launch_bounds__(256) k_gemm_xW(const float* __restrict__ x,
                                                 const _Float16* __restrict__ WT,
                                                 float* __restrict__ h) {
  __shared__ _Float16 sA[16 * DIM];          //  8 KB
  __shared__ _Float16 sB[2][DIM * 32];       // 32 KB (double-buffered B slice)
  int tid = threadIdx.x;
  int nb = blockIdx.x * 16;

  // stage x tile as f16 A matrix
  for (int i = tid; i < 16 * DIM; i += 256) {
    int r = i >> 8, c = i & 255;
    sA[i] = (_Float16)x[(size_t)(nb + r) * DIM + c];
  }

  // prologue: async-prefetch B slice 0 (each thread copies 4x16B chunks)
  {
#pragma unroll
    for (int i2 = 0; i2 < 4; ++i2) {
      int chunk = tid + i2 * 256;          // 1024 chunks of 16B = 16 KB
      int n = chunk >> 2, part = chunk & 3;
      async_copy_b128(&sB[0][n * 32 + part * 8], WT + (size_t)n * DIM + part * 8);
    }
  }

  int wave = tid >> 5, lane = tid & 31;
  int n0 = wave * 32;
  v8f acc0 = {}, acc1 = {};

#pragma unroll
  for (int kt = 0; kt < 8; ++kt) {
    wait_asynccnt0();        // own wave's async slice-kt loads complete
    __syncthreads();         // all waves' loads (and sA stores on kt==0) visible
    if (kt < 7) {            // prefetch next slice into the other buffer
      int k0n = (kt + 1) * 32;
#pragma unroll
      for (int i2 = 0; i2 < 4; ++i2) {
        int chunk = tid + i2 * 256;
        int n = chunk >> 2, part = chunk & 3;
        async_copy_b128(&sB[(kt + 1) & 1][n * 32 + part * 8],
                        WT + (size_t)n * DIM + k0n + part * 8);
      }
    }
    const _Float16* bcur = &sB[kt & 1][0];
    v16h a  = wmma_load_a(sA + kt * 32, DIM, lane);
    v16h b0 = wmma_load_b(bcur, 32, n0,      0, lane);
    v16h b1 = wmma_load_b(bcur, 32, n0 + 16, 0, lane);
    acc0 = wmma16(a, b0, acc0);
    acc1 = wmma16(a, b1, acc1);
  }

  int col = lane & 15, hi = lane >> 4;
#pragma unroll
  for (int r = 0; r < 8; ++r) {
    int row = r + hi * 8;
    h[(size_t)(nb + row) * DIM + n0 + col]      = acc0[r];
    h[(size_t)(nb + row) * DIM + n0 + 16 + col] = acc1[r];
  }
}

// ---------------------------------------------------------------------------
// K2: alpha_src/alpha_dst [N,H] — one wave per (node,head)
// ---------------------------------------------------------------------------
__global__ void k_alphas(const float* __restrict__ h,
                         const float* __restrict__ a_src, const float* __restrict__ a_dst,
                         float* __restrict__ asrc, float* __restrict__ adst) {
  int gw = blockIdx.x * 8 + (threadIdx.x >> 5);
  if (gw >= NN * HEADS) return;
  int lane = threadIdx.x & 31;
  int n = gw >> 2, hd = gw & 3;
  const float* hp = h + (size_t)n * DIM + hd * HDIM;
  const float* ws = a_src + hd * HDIM;
  const float* wd = a_dst + hd * HDIM;
  float s = hp[lane] * ws[lane] + hp[lane + 32] * ws[lane + 32];
  float d = hp[lane] * wd[lane] + hp[lane + 32] * wd[lane + 32];
#pragma unroll
  for (int off = 16; off > 0; off >>= 1) {
    s += __shfl_xor(s, off, 32);
    d += __shfl_xor(d, off, 32);
  }
  if (lane == 0) { asrc[gw] = s; adst[gw] = d; }
}

// ---------------------------------------------------------------------------
// K3: edge pass 1 — segment max of leaky_relu logits (atomic float max)
// ---------------------------------------------------------------------------
__global__ void k_edge_max(const long long* __restrict__ ei,
                           const float* __restrict__ asrc, const float* __restrict__ adst,
                           float* __restrict__ mbuf) {
  int gid = blockIdx.x * blockDim.x + threadIdx.x;
  if (gid >= EE * HEADS) return;
  int e = gid >> 2, hd = gid & 3;
  long long s = ei[e], d = ei[EE + e];
  float ev = asrc[s * HEADS + hd] + adst[d * HEADS + hd];
  ev = (ev > 0.0f) ? ev : NEG_SLOPE * ev;
  atomicMaxF(&mbuf[d * HEADS + hd], ev);
}

// ---------------------------------------------------------------------------
// K4: edge pass 2 — ex = exp(e - m[dst]); denom += ex (atomic)
// ---------------------------------------------------------------------------
__global__ void k_edge_exp(const long long* __restrict__ ei,
                           const float* __restrict__ asrc, const float* __restrict__ adst,
                           const float* __restrict__ mbuf,
                           float* __restrict__ exbuf, float* __restrict__ denom) {
  int gid = blockIdx.x * blockDim.x + threadIdx.x;
  if (gid >= EE * HEADS) return;
  int e = gid >> 2, hd = gid & 3;
  long long s = ei[e], d = ei[EE + e];
  float ev = asrc[s * HEADS + hd] + adst[d * HEADS + hd];
  ev = (ev > 0.0f) ? ev : NEG_SLOPE * ev;
  float ex = __expf(ev - mbuf[d * HEADS + hd]);
  exbuf[(size_t)e * HEADS + hd] = ex;
  atomicAdd(&denom[d * HEADS + hd], ex);
}

// ---------------------------------------------------------------------------
// K5: edge pass 3 — acc[dst] += h[src] * alpha  (one wave per edge,
//     8 channels per lane; head = lane>>3)
// ---------------------------------------------------------------------------
__global__ void k_edge_scatter(const long long* __restrict__ ei,
                               const float* __restrict__ h,
                               const float* __restrict__ exbuf,
                               const float* __restrict__ denom,
                               float* __restrict__ acc) {
  int e = blockIdx.x * 8 + (threadIdx.x >> 5);
  if (e >= EE) return;
  int lane = threadIdx.x & 31;
  long long s = ei[e], d = ei[EE + e];
  int c0 = lane * 8;
  int hd = lane >> 3;
  float alpha = exbuf[(size_t)e * HEADS + hd] / (denom[d * HEADS + hd] + 1e-16f);
  const float4* hv = (const float4*)(h + (size_t)s * DIM + c0);
  float4 v0 = hv[0], v1 = hv[1];
  float* ap = acc + (size_t)d * DIM + c0;
  atomicAdd(ap + 0, v0.x * alpha);
  atomicAdd(ap + 1, v0.y * alpha);
  atomicAdd(ap + 2, v0.z * alpha);
  atomicAdd(ap + 3, v0.w * alpha);
  atomicAdd(ap + 4, v1.x * alpha);
  atomicAdd(ap + 5, v1.y * alpha);
  atomicAdd(ap + 6, v1.z * alpha);
  atomicAdd(ap + 7, v1.w * alpha);
}

// ---------------------------------------------------------------------------
// K6: x1 = LayerNorm(x + acc + gat_b)   (one block per node)
// ---------------------------------------------------------------------------
__global__ void __launch_bounds__(256) k_ln1(const float* __restrict__ x,
                                             const float* __restrict__ acc,
                                             const float* __restrict__ gat_b,
                                             const float* __restrict__ g,
                                             const float* __restrict__ b,
                                             float* __restrict__ x1) {
  __shared__ float s1[256], s2[256];
  int n = blockIdx.x, c = threadIdx.x;
  float y = x[(size_t)n * DIM + c] + acc[(size_t)n * DIM + c] + gat_b[c];
  s1[c] = y; s2[c] = y * y;
  __syncthreads();
#pragma unroll
  for (int off = 128; off > 0; off >>= 1) {
    if (c < off) { s1[c] += s1[c + off]; s2[c] += s2[c + off]; }
    __syncthreads();
  }
  float mu = s1[0] * (1.0f / DIM);
  float var = s2[0] * (1.0f / DIM) - mu * mu;
  float rs = rsqrtf(var + LN_EPS);
  x1[(size_t)n * DIM + c] = (y - mu) * rs * g[c] + b[c];
}

// ---------------------------------------------------------------------------
// K7: fused FFN + LN2 + residual (16 rows per block, 8 waves)
//   hid = gelu(x1@W1 + b1)    [16,1024]  — WMMA, staged f16 in LDS
//   y2  = x1 + hid@W2 + b2    [16,256]
//   out = LN(y2)*g + b + x_orig
// ---------------------------------------------------------------------------
__global__ void __launch_bounds__(256) k_ffn(const float* __restrict__ x1,
                                             const float* __restrict__ x0,
                                             const _Float16* __restrict__ W1T,
                                             const float* __restrict__ b1,
                                             const _Float16* __restrict__ W2T,
                                             const float* __restrict__ b2,
                                             const float* __restrict__ g,
                                             const float* __restrict__ bta,
                                             float* __restrict__ out) {
  __shared__ _Float16 sA[16 * DIM];     //  8 KB
  __shared__ _Float16 sH[16 * FFD];     // 32 KB
  __shared__ float    sY[16 * DIM];     // 16 KB
  __shared__ float    sSum[16][16], sSq[16][16];
  __shared__ float    sStat[16][2];

  int tid = threadIdx.x;
  int nb = blockIdx.x * 16;
  int wave = tid >> 5, lane = tid & 31;
  int col = lane & 15, hi = lane >> 4;

  // stage x1 tile as f16 A matrix
  for (int i = tid; i < 16 * DIM; i += 256) {
    int r = i >> 8, c = i & 255;
    sA[i] = (_Float16)x1[(size_t)(nb + r) * DIM + c];
  }
  __syncthreads();

  // GEMM1: wave covers hidden columns [wave*128, wave*128+128)
  v8f zero = {};
  v8f acc[8];
#pragma unroll
  for (int t = 0; t < 8; ++t) acc[t] = zero;
#pragma unroll
  for (int kt = 0; kt < 8; ++kt) {
    int k0 = kt * 32;
    v16h a = wmma_load_a(sA + k0, DIM, lane);
#pragma unroll
    for (int t = 0; t < 8; ++t) {
      v16h bfr = wmma_load_b(W1T, DIM, wave * 128 + t * 16, k0, lane);
      acc[t] = wmma16(a, bfr, acc[t]);
    }
  }
  // bias + GELU -> f16 hidden in LDS
#pragma unroll
  for (int t = 0; t < 8; ++t) {
    int n = wave * 128 + t * 16 + col;
#pragma unroll
    for (int r = 0; r < 8; ++r) {
      int row = r + hi * 8;
      float v = acc[t][r] + b1[n];
      sH[row * FFD + n] = (_Float16)gelu_exact(v);
    }
  }
  __syncthreads();

  // GEMM2: wave covers output columns [wave*32, wave*32+32), K = 1024
  v8f o0 = zero, o1 = zero;
#pragma unroll 4
  for (int kt = 0; kt < 32; ++kt) {
    int k0 = kt * 32;
    v16h a  = wmma_load_a(sH + k0, FFD, lane);
    v16h b0 = wmma_load_b(W2T, FFD, wave * 32,      k0, lane);
    v16h b1f = wmma_load_b(W2T, FFD, wave * 32 + 16, k0, lane);
    o0 = wmma16(a, b0, o0);
    o1 = wmma16(a, b1f, o1);
  }
  // y2 = x1 + ffn + b2 -> LDS
#pragma unroll
  for (int r = 0; r < 8; ++r) {
    int row = r + hi * 8;
    int c0 = wave * 32 + col;
    int c1 = c0 + 16;
    sY[row * DIM + c0] = x1[(size_t)(nb + row) * DIM + c0] + o0[r] + b2[c0];
    sY[row * DIM + c1] = x1[(size_t)(nb + row) * DIM + c1] + o1[r] + b2[c1];
  }
  __syncthreads();

  // LN stats: 16 threads per row, 16 elems each
  {
    int row = tid >> 4, seg = tid & 15;
    float s = 0.0f, q = 0.0f;
#pragma unroll
    for (int j = 0; j < 16; ++j) {
      float v = sY[row * DIM + seg * 16 + j];
      s += v; q += v * v;
    }
    sSum[row][seg] = s; sSq[row][seg] = q;
  }
  __syncthreads();
  if (tid < 16) {
    float s = 0.0f, q = 0.0f;
#pragma unroll
    for (int j = 0; j < 16; ++j) { s += sSum[tid][j]; q += sSq[tid][j]; }
    float mu = s * (1.0f / DIM);
    float var = q * (1.0f / DIM) - mu * mu;
    sStat[tid][0] = mu;
    sStat[tid][1] = rsqrtf(var + LN_EPS);
  }
  __syncthreads();

  // out = LN(y2)*g + b + residual(x0)
  {
    int c = tid;  // 0..255
    float gc = g[c], bc = bta[c];
#pragma unroll
    for (int row = 0; row < 16; ++row) {
      float v = (sY[row * DIM + c] - sStat[row][0]) * sStat[row][1] * gc + bc;
      out[(size_t)(nb + row) * DIM + c] = v + x0[(size_t)(nb + row) * DIM + c];
    }
  }
}

// ---------------------------------------------------------------------------
extern "C" void kernel_launch(void* const* d_in, const int* in_sizes, int n_in,
                              void* d_out, int out_size, void* d_ws, size_t ws_size,
                              hipStream_t stream) {
  const float*      x     = (const float*)d_in[0];
  const long long*  ei    = (const long long*)d_in[1];
  const float*      W     = (const float*)d_in[2];
  const float*      a_src = (const float*)d_in[3];
  const float*      a_dst = (const float*)d_in[4];
  const float*      gat_b = (const float*)d_in[5];
  const float*      ln_g  = (const float*)d_in[6];
  const float*      ln_b  = (const float*)d_in[7];
  const float*      W1    = (const float*)d_in[8];
  const float*      b1    = (const float*)d_in[9];
  const float*      W2    = (const float*)d_in[10];
  const float*      b2    = (const float*)d_in[11];
  float* out = (float*)d_out;

  unsigned char* ws = (unsigned char*)d_ws;
  size_t o = 0;
  auto alloc = [&](size_t bytes) -> void* {
    void* p = ws + o;
    o += (bytes + 255) & ~(size_t)255;
    return p;
  };
  float*     h      = (float*)alloc((size_t)NN * DIM * 4);
  float*     x1     = (float*)alloc((size_t)NN * DIM * 4);
  float*     acc    = (float*)alloc((size_t)NN * DIM * 4);
  float*     asrc   = (float*)alloc((size_t)NN * HEADS * 4);
  float*     adst   = (float*)alloc((size_t)NN * HEADS * 4);
  float*     mbuf   = (float*)alloc((size_t)NN * HEADS * 4);
  float*     denom  = (float*)alloc((size_t)NN * HEADS * 4);
  float*     exbuf  = (float*)alloc((size_t)EE * HEADS * 4);
  _Float16*  WT     = (_Float16*)alloc((size_t)DIM * DIM * 2);
  _Float16*  W1T    = (_Float16*)alloc((size_t)DIM * FFD * 2);
  _Float16*  W2T    = (_Float16*)alloc((size_t)FFD * DIM * 2);
  (void)ws_size; (void)in_sizes; (void)n_in; (void)out_size;

  // weight conversion (transposed f16)
  k_convert_wT<<<(DIM * DIM + 255) / 256, 256, 0, stream>>>(W,  WT,  DIM, DIM);
  k_convert_wT<<<(DIM * FFD + 255) / 256, 256, 0, stream>>>(W1, W1T, DIM, FFD);
  k_convert_wT<<<(FFD * DIM + 255) / 256, 256, 0, stream>>>(W2, W2T, FFD, DIM);

  // init accumulators
  k_init<<<4096, 256, 0, stream>>>(acc, denom, mbuf);

  // h = x @ W   (WMMA + async double-buffered B staging)
  k_gemm_xW<<<NN / 16, 256, 0, stream>>>(x, WT, h);

  // attention logits per (node, head)
  k_alphas<<<(NN * HEADS + 7) / 8, 256, 0, stream>>>(h, a_src, a_dst, asrc, adst);

  // segment softmax over edges
  k_edge_max<<<(EE * HEADS + 255) / 256, 256, 0, stream>>>(ei, asrc, adst, mbuf);
  k_edge_exp<<<(EE * HEADS + 255) / 256, 256, 0, stream>>>(ei, asrc, adst, mbuf, exbuf, denom);
  k_edge_scatter<<<(EE + 7) / 8, 256, 0, stream>>>(ei, h, exbuf, denom, acc);

  // x1 = LN(x + gat_out)
  k_ln1<<<NN, 256, 0, stream>>>(x, acc, gat_b, ln_g, ln_b, x1);

  // fused FFN + LN2 + residual
  k_ffn<<<NN / 16, 256, 0, stream>>>(x1, x, W1T, b1, W2T, b2, ln_g, ln_b, out);
}
